// TaskInfoConv_65163243815762
// MI455X (gfx1250) — matrix operand order repaired and checked by reading, not compiled, not run on hardware
//
#include <hip/hip_runtime.h>
#include <hip/hip_bf16.h>
#include <math.h>

#define N_TASK 50000
#define N_RET  20000
#define N_DRAM 5000
#define N_LINK 100000
#define E_RET  1000000
#define E_DRAM 250000
#define E_LINK 2000000
#define HDIM   64
#define K3H    192

typedef __attribute__((ext_vector_type(16))) _Float16 v16h;
typedef __attribute__((ext_vector_type(8)))  float    v8f;

// ---- hardware tanh (CDNA5 V_TANH_F32) ----
__device__ __forceinline__ float fast_tanh(float x) {
#if __has_builtin(__builtin_amdgcn_tanhf)
  return __builtin_amdgcn_tanhf(x);   // compiler manages TRANS32 hazards
#else
  float y;
  // v_nop satisfies the trans-op result hazard (ISA: 1 indep op / V_NOP
  // after a transcendental before its output may be consumed).
  asm volatile("v_tanh_f32 %0, %1\n\tv_nop" : "=v"(y) : "v"(x));
  return y;
#endif
}

// ---- monotone float <-> unsigned mapping for atomic max/min on floats ----
__device__ __forceinline__ unsigned f2ord(float f) {
  unsigned u = __float_as_uint(f);
  return (u & 0x80000000u) ? ~u : (u | 0x80000000u);
}
__device__ __forceinline__ float ord2f(unsigned u) {
  return (u & 0x80000000u) ? __uint_as_float(u & 0x7FFFFFFFu)
                           : __uint_as_float(~u);
}

// ---------------- init ----------------
__global__ void fill_u32_kernel(unsigned* __restrict__ p, unsigned val, long n) {
  long i = (long)blockIdx.x * blockDim.x + threadIdx.x;
  long stride = (long)gridDim.x * blockDim.x;
  for (; i < n; i += stride) p[i] = val;
}

// ---------------- s[mod] += feat ----------------
__global__ void scatter_add_kernel(const float* __restrict__ feat,
                                   const int* __restrict__ mod,
                                   int E, float* __restrict__ s) {
  int i = blockIdx.x * blockDim.x + threadIdx.x;
  int stride = gridDim.x * blockDim.x;
  for (; i < E; i += stride) atomicAdd(&s[mod[i]], feat[i]);
}

// ---------------- per-task scalar min/max of s[mod] ----------------
__global__ void edge_minmax_kernel(const float* __restrict__ s,
                                   const int* __restrict__ mod,
                                   const int* __restrict__ task,
                                   int E,
                                   unsigned* __restrict__ smax,
                                   unsigned* __restrict__ smin) {
  int i = blockIdx.x * blockDim.x + threadIdx.x;
  int stride = gridDim.x * blockDim.x;
  for (; i < E; i += stride) {
    unsigned u = f2ord(s[mod[i]]);
    int t = task[i];
    atomicMax(&smax[t], u);  // sentinel 0 is below any real value's ordering
    atomicMin(&smin[t], u);
  }
}

// ---------------- h_task = tanh(h_cat @ W_task^T + b) via WMMA ----------------
// One wave32 per 16-task tile; 4 column tiles of 16; K=192 in 6 chunks of 32.
// B is staged in LDS in *fragment order*: for (chunk c, col-tile nt, lane),
// the lane's 16 halves are contiguous & 32B aligned -> two ds_load_b128.
__global__ __launch_bounds__(256) void task_wmma_kernel(
    const unsigned* __restrict__ smax_ord,  // [3][N_TASK] ordered-uint
    const unsigned* __restrict__ smin_ord,  // [3][N_TASK]
    const float* __restrict__ W_ret,  const float* __restrict__ b_ret,
    const float* __restrict__ W_dram, const float* __restrict__ b_dram,
    const float* __restrict__ W_link, const float* __restrict__ b_link,
    const float* __restrict__ W_task, const float* __restrict__ b_task,
    float* __restrict__ h_task) {
  // sB[((c*4 + nt)*32 + lane)*16 + j] = W_task[n][k]
  //   with row = lane&15, hi = lane>>4, n = nt*16+row, k = c*32 + hi*16 + j
  __shared__ __align__(32) _Float16 sB[6 * 4 * 32 * 16];  // 24 KB
  for (int idx = threadIdx.x; idx < 6 * 4 * 32 * 16; idx += blockDim.x) {
    const int j  = idx & 15;
    const int ls = (idx >> 4) & 31;
    const int cn = idx >> 9;
    const int c  = cn >> 2;
    const int nt = cn & 3;
    const int k  = c * 32 + ((ls >> 4) << 4) + j;
    const int n  = nt * 16 + (ls & 15);
    sB[idx] = (_Float16)W_task[n * K3H + k];
  }
  __syncthreads();

  const int lane = threadIdx.x & 31;
  const int wave = threadIdx.x >> 5;
  const int ntiles = (N_TASK + 15) >> 4;  // 3125
  int tile = blockIdx.x * 8 + wave;
  if (tile >= ntiles) tile = ntiles - 1;  // duplicate waves rewrite identical data
  const int tbase = tile * 16;

  const int row = lane & 15;   // A-matrix M for this lane; also B/C column-in-group
  const int hi  = lane >> 4;   // lane half: K-group select / C row offset
  const int t   = tbase + row;

  // per-etype max/min of s over this lane's task row (branchless decode)
  float vmx[3], vmn[3];
  bool has[3];
#pragma unroll
  for (int et = 0; et < 3; ++et) {
    const unsigned mo = smax_ord[et * N_TASK + t];
    has[et] = (mo != 0u);                  // zero in-degree (per etype) -> 0
    vmx[et] = ord2f(mo);
    vmn[et] = ord2f(smin_ord[et * N_TASK + t]);
  }

  v8f acc[4] = {v8f{}, v8f{}, v8f{}, v8f{}};

  const float* Wes[3] = {W_ret, W_dram, W_link};
  const float* bes[3] = {b_ret, b_dram, b_link};

#pragma unroll
  for (int c = 0; c < 6; ++c) {          // K chunk of 32; etype = c/2
    const int et = c >> 1;
    const float* We = Wes[et];
    const float* be = bes[et];
    const bool hs = has[et];
    const float vmax = vmx[et];
    const float vmin = vmn[et];

    // A fragment (contiguous k order, same as sB): lanes 0-15 hold K {0..15},
    // lanes 16-31 hold K {16..31}, row M = lane&15.
    v16h a;
#pragma unroll
    for (int j = 0; j < 16; ++j) {
      const int kk = hi * 16 + j;
      const int h = ((c & 1) << 5) + kk;        // feature index within etype
      const float w = We[h];
      float x = fmaf(w, (w > 0.f ? vmax : vmin), be[h]);
      // branchless zero-degree handling: tanh(0) == 0
      // (can't scale the output: sentinel decodes to NaN and NaN*0 != 0)
      x = hs ? x : 0.0f;
      a[j] = (_Float16)fast_tanh(x);
    }

    // 4 column tiles of 16: one 32-byte vector LDS load per fragment
#pragma unroll
    for (int nt = 0; nt < 4; ++nt) {
      const v16h b = *(const v16h*)&sB[(((c * 4 + nt) * 32) + lane) * 16];
      acc[nt] = __builtin_amdgcn_wmma_f32_16x16x32_f16(
          /*neg_a=*/false, a, /*neg_b=*/false, b,
          /*c_mod=*/(short)0, acc[nt], /*reuse_a=*/false, /*reuse_b=*/false);
    }
  }

  // C/D layout: VGPR r, lanes 0-15 -> M=r, lanes 16-31 -> M=r+8; N = lane&15
#pragma unroll
  for (int nt = 0; nt < 4; ++nt) {
    const int n = nt * 16 + row;
    const float bias = b_task[n];
#pragma unroll
    for (int r = 0; r < 8; ++r) {
      const int M = r + hi * 8;
      h_task[(size_t)(tbase + M) * HDIM + n] = fast_tanh(acc[nt][r] + bias);
    }
  }
}

// ---------------- mean over link edges ----------------
__global__ void link_accum_kernel(const float* __restrict__ h_task,
                                  const int* __restrict__ task_link,
                                  const int* __restrict__ mod_link,
                                  float* __restrict__ out,
                                  float* __restrict__ deg) {
  const long total = (long)E_LINK * HDIM;
  long i = (long)blockIdx.x * blockDim.x + threadIdx.x;
  long stride = (long)gridDim.x * blockDim.x;
  for (; i < total; i += stride) {
    const int e = (int)(i >> 6);
    const int h = (int)(i & 63);
    const int t = task_link[e];
    const int m = mod_link[e];
    atomicAdd(&out[(size_t)m * HDIM + h], h_task[(size_t)t * HDIM + h]);
    if (h == 0) atomicAdd(&deg[m], 1.0f);
  }
}

__global__ void finalize_kernel(const float* __restrict__ deg,
                                float* __restrict__ out) {
  const long total = (long)N_LINK * HDIM;
  long i = (long)blockIdx.x * blockDim.x + threadIdx.x;
  long stride = (long)gridDim.x * blockDim.x;
  for (; i < total; i += stride) {
    const float d = deg[i >> 6];
    out[i] = (d > 0.f) ? out[i] * (1.f / d) : 0.f;
  }
}

extern "C" void kernel_launch(void* const* d_in, const int* in_sizes, int n_in,
                              void* d_out, int out_size, void* d_ws, size_t ws_size,
                              hipStream_t stream) {
  const float* feat_ret  = (const float*)d_in[0];
  const float* feat_dram = (const float*)d_in[1];
  const float* feat_link = (const float*)d_in[2];
  const float* W_ret  = (const float*)d_in[3];
  const float* b_ret  = (const float*)d_in[4];
  const float* W_dram = (const float*)d_in[5];
  const float* b_dram = (const float*)d_in[6];
  const float* W_link = (const float*)d_in[7];
  const float* b_link = (const float*)d_in[8];
  const float* W_task = (const float*)d_in[9];
  const float* b_task = (const float*)d_in[10];
  const int* task_ret  = (const int*)d_in[11];
  const int* mod_ret   = (const int*)d_in[12];
  const int* task_dram = (const int*)d_in[13];
  const int* mod_dram  = (const int*)d_in[14];
  const int* task_link = (const int*)d_in[15];
  const int* mod_link  = (const int*)d_in[16];
  float* out = (float*)d_out;

  // workspace layout (4-byte words), ~14.9 MB total
  float* ws = (float*)d_ws;
  float*    s_ret  = ws;                              // 20000
  float*    s_dram = ws + 20000;                      // 5000
  float*    s_link = ws + 25000;                      // 100000
  float*    deg    = ws + 125000;                     // 100000
  unsigned* smax   = (unsigned*)(ws + 225000);        // 3*N_TASK
  unsigned* smin   = (unsigned*)(ws + 375000);        // 3*N_TASK
  float*    h_task = ws + 525000;                     // N_TASK*HDIM

  // init: zero [s_ret|s_dram|s_link|deg|smax] (375000 words), smin=0xFFFFFFFF, out=0
  fill_u32_kernel<<<(375000 + 255) / 256, 256, 0, stream>>>((unsigned*)ws, 0u, 375000);
  fill_u32_kernel<<<(3 * N_TASK + 255) / 256, 256, 0, stream>>>(smin, 0xFFFFFFFFu, 3 * N_TASK);
  fill_u32_kernel<<<((long)N_LINK * HDIM + 255) / 256, 256, 0, stream>>>(
      (unsigned*)out, 0u, (long)N_LINK * HDIM);

  // per-module segment sums
  scatter_add_kernel<<<(E_RET  + 255) / 256, 256, 0, stream>>>(feat_ret,  mod_ret,  E_RET,  s_ret);
  scatter_add_kernel<<<(E_DRAM + 255) / 256, 256, 0, stream>>>(feat_dram, mod_dram, E_DRAM, s_dram);
  scatter_add_kernel<<<(E_LINK + 255) / 256, 256, 0, stream>>>(feat_link, mod_link, E_LINK, s_link);

  // per-task scalar min/max (replaces H-wide segment-max; tanh monotone)
  edge_minmax_kernel<<<(E_RET  + 255) / 256, 256, 0, stream>>>(s_ret,  mod_ret,  task_ret,  E_RET,
                                                               smax + 0 * N_TASK, smin + 0 * N_TASK);
  edge_minmax_kernel<<<(E_DRAM + 255) / 256, 256, 0, stream>>>(s_dram, mod_dram, task_dram, E_DRAM,
                                                               smax + 1 * N_TASK, smin + 1 * N_TASK);
  edge_minmax_kernel<<<(E_LINK + 255) / 256, 256, 0, stream>>>(s_link, mod_link, task_link, E_LINK,
                                                               smax + 2 * N_TASK, smin + 2 * N_TASK);

  // task GEMM via WMMA: 3125 tiles of 16 tasks, 8 waves/block
  const int ntiles = (N_TASK + 15) / 16;
  const int nblocks = (ntiles + 7) / 8;
  task_wmma_kernel<<<nblocks, 256, 0, stream>>>(smax, smin, W_ret, b_ret, W_dram, b_dram,
                                                W_link, b_link, W_task, b_task, h_task);

  // mean over link edges
  const long tot = (long)E_LINK * HDIM;
  link_accum_kernel<<<(int)((tot + 255) / 256), 256, 0, stream>>>(h_task, task_link, mod_link, out, deg);
  finalize_kernel<<<((long)N_LINK * HDIM + 255) / 256, 256, 0, stream>>>(deg, out);
}
